// LSTMModel_8710193676360
// MI455X (gfx1250) — matrix operand order repaired and checked by reading, not compiled, not run on hardware
//
#include <hip/hip_runtime.h>
#include <hip/hip_bf16.h>

typedef __bf16 bf16_t;
typedef __attribute__((ext_vector_type(16))) __bf16 v16bf;
typedef __attribute__((ext_vector_type(8)))  float  v8f;

constexpr int B  = 64;
constexpr int T  = 512;
constexpr int D  = 1024;
constexpr int H  = 1024;
constexpr int L  = 4;
constexpr int O  = 256;
constexpr int FOURH = 4 * H;

// ---------------------------------------------------------------------------
// Fragment loaders for 16-bit WMMA: this code's contiguous k order, same for A and B
//
// A (16x32, MxK):  lane m = l%16; lanes 0-15 hold K 0..15,
//                  lanes 16-31 hold K 16..31
//                  -> one contiguous 32B run (two 16B loads).
// B (32x16, KxN):  lane col n = l%16; lanes 0-15 hold K 0..15,
//                  lanes 16-31 hold K 16..31
//                  -> one contiguous 32B run (two 16B loads).
// ---------------------------------------------------------------------------
static __device__ __forceinline__ v16bf ld_frag(const bf16_t* p0, const bf16_t* p1) {
    union U { v16bf v; uint4 q[2]; } u;
    u.q[0] = *reinterpret_cast<const uint4*>(p0);
    u.q[1] = *reinterpret_cast<const uint4*>(p1);
    return u.v;
}

static __device__ __forceinline__ v8f wmma_bf16(v16bf a, v16bf b, v8f c) {
    // (neg_a, A, neg_b, B, c_mod, C, reuse_a, reuse_b)
    return __builtin_amdgcn_wmma_f32_16x16x32_bf16(false, a, false, b, (short)0, c,
                                                   false, false);
}

// ---------------------------------------------------------------------------
// Fused LSTM step for one (t, layer): gates = [inA@Wih^T] + hprev@Whh^T + bsum
// (+ optional precomputed input-projection xgate), then cell update.
// Grid: 32 blocks x 512 threads (16 waves).
// Block b owns h-columns [32b, 32b+32) across all 4 gates and all 64 batch rows.
// Wave w: M-tile (w>>2), gate (w&3), two 16-col N tiles.
// ---------------------------------------------------------------------------
__global__ __launch_bounds__(512)
void lstm_step_kernel(const bf16_t* __restrict__ inA, unsigned long long ldA,
                      const float*  __restrict__ xgate,   // [B,4H] or nullptr
                      const bf16_t* __restrict__ hprev,
                      const bf16_t* __restrict__ Wih,
                      const bf16_t* __restrict__ Whh,
                      const float*  __restrict__ bsum,
                      float*        __restrict__ cst,
                      bf16_t*       __restrict__ hout)
{
    __shared__ float gl[4][B][32];

    const int lane = threadIdx.x & 31;
    const int w    = threadIdx.x >> 5;
    const int mt   = w >> 2;          // 0..3   (16 batch rows each)
    const int g    = w & 3;           // gate id: 0=i 1=f 2=g 3=o
    const int J    = blockIdx.x * 32; // h-column base

    const int m   = lane & 15;
    const int kA0 = (lane < 16) ? 0 : 16;
    const int nB  = lane & 15;
    const int kB0 = (lane < 16) ? 0 : 16;

    v8f acc0 = {};
    v8f acc1 = {};

    // ---- input projection: inA [64 x D] (row stride ldA) @ Wih^T ----
    // Skipped (uniform branch, EXEC stays all-ones) when precomputed.
    if (xgate == nullptr) {
        const bf16_t* arow = inA + (unsigned long long)(mt * 16 + m) * ldA;
        const bf16_t* b0r  = Wih + (size_t)(g * H + J + nB) * D;
        const bf16_t* b1r  = Wih + (size_t)(g * H + J + 16 + nB) * D;
        for (int kk = 0; kk < D; kk += 32) {
            __builtin_prefetch(b0r + kk + 256, 0, 1);  // global_prefetch_b8
            v16bf a  = ld_frag(arow + kk + kA0, arow + kk + kA0 + 8);
            v16bf b0 = ld_frag(b0r  + kk + kB0, b0r  + kk + kB0 + 8);
            v16bf b1 = ld_frag(b1r  + kk + kB0, b1r  + kk + kB0 + 8);
            acc0 = wmma_bf16(a, b0, acc0);
            acc1 = wmma_bf16(a, b1, acc1);
        }
    }
    // ---- recurrent projection: hprev [64 x H] @ Whh^T ----
    {
        const bf16_t* arow = hprev + (size_t)(mt * 16 + m) * H;
        const bf16_t* b0r  = Whh + (size_t)(g * H + J + nB) * H;
        const bf16_t* b1r  = Whh + (size_t)(g * H + J + 16 + nB) * H;
        for (int kk = 0; kk < H; kk += 32) {
            __builtin_prefetch(b0r + kk + 256, 0, 1);
            v16bf a  = ld_frag(arow + kk + kA0, arow + kk + kA0 + 8);
            v16bf b0 = ld_frag(b0r  + kk + kB0, b0r  + kk + kB0 + 8);
            v16bf b1 = ld_frag(b1r  + kk + kB0, b1r  + kk + kB0 + 8);
            acc0 = wmma_bf16(a, b0, acc0);
            acc1 = wmma_bf16(a, b1, acc1);
        }
    }

    // C/D layout: lanes 0-15 -> N=lane, M=r; lanes 16-31 -> N=lane-16, M=8+r
    const int mo = (lane < 16) ? 0 : 8;
#pragma unroll
    for (int r = 0; r < 8; ++r) {
        gl[g][mt * 16 + mo + r][nB]      = acc0[r];
        gl[g][mt * 16 + mo + r][16 + nB] = acc1[r];
    }
    __syncthreads();

    // ---- cell update for the 64x32 slice this block owns ----
    for (int i = threadIdx.x; i < B * 32; i += 512) {
        const int b  = i >> 5;
        const int jj = i & 31;
        const int j  = J + jj;
        float ig = gl[0][b][jj] + bsum[0 * H + j];
        float fg = gl[1][b][jj] + bsum[1 * H + j];
        float gg = gl[2][b][jj] + bsum[2 * H + j];
        float og = gl[3][b][jj] + bsum[3 * H + j];
        if (xgate != nullptr) {
            const float* xb = xgate + (size_t)b * FOURH;
            ig += xb[0 * H + j];
            fg += xb[1 * H + j];
            gg += xb[2 * H + j];
            og += xb[3 * H + j];
        }
        float si = 1.0f / (1.0f + __expf(-ig));
        float sf = 1.0f / (1.0f + __expf(-fg));
        float so = 1.0f / (1.0f + __expf(-og));
        float tg = tanhf(gg);
        const size_t idx = (size_t)b * H + j;
        float cn = sf * cst[idx] + si * tg;
        float hn = so * tanhf(cn);
        cst[idx]  = cn;
        hout[idx] = (bf16_t)hn;
    }
}

// ---------------------------------------------------------------------------
// Hoisted layer-0 input projection over ALL timesteps (no serial dependence):
// xg[t][b][4H] = x[b][t][:] @ Wih0^T.  Grid: (32 col-blocks, T), 512 threads.
// Fully parallel throughput GEMM: 275 GFLOP across 16384 blocks.
// ---------------------------------------------------------------------------
__global__ __launch_bounds__(512)
void xproj_kernel(const bf16_t* __restrict__ xbf,   // [B,T,D]
                  const bf16_t* __restrict__ Wih0,  // [4H,D]
                  float*        __restrict__ xg)    // [T,B,4H]
{
    const int lane = threadIdx.x & 31;
    const int w    = threadIdx.x >> 5;
    const int mt   = w >> 2;
    const int g    = w & 3;
    const int J    = blockIdx.x * 32;
    const int t    = blockIdx.y;

    const int m   = lane & 15;
    const int kA0 = (lane < 16) ? 0 : 16;
    const int nB  = lane & 15;
    const int kB0 = (lane < 16) ? 0 : 16;

    v8f acc0 = {};
    v8f acc1 = {};

    const bf16_t* arow = xbf + (size_t)(mt * 16 + m) * T * D + (size_t)t * D;
    const bf16_t* b0r  = Wih0 + (size_t)(g * H + J + nB) * D;
    const bf16_t* b1r  = Wih0 + (size_t)(g * H + J + 16 + nB) * D;
    for (int kk = 0; kk < D; kk += 32) {
        __builtin_prefetch(b0r + kk + 256, 0, 1);
        v16bf a  = ld_frag(arow + kk + kA0, arow + kk + kA0 + 8);
        v16bf b0 = ld_frag(b0r  + kk + kB0, b0r  + kk + kB0 + 8);
        v16bf b1 = ld_frag(b1r  + kk + kB0, b1r  + kk + kB0 + 8);
        acc0 = wmma_bf16(a, b0, acc0);
        acc1 = wmma_bf16(a, b1, acc1);
    }

    const int mo = (lane < 16) ? 0 : 8;
    float* outt = xg + (size_t)t * B * FOURH;
#pragma unroll
    for (int r = 0; r < 8; ++r) {
        const size_t row = (size_t)(mt * 16 + mo + r) * FOURH + g * H + J;
        outt[row + nB]      = acc0[r];
        outt[row + 16 + nB] = acc1[r];
    }
}

// ---------------------------------------------------------------------------
// Final FC: out[64,256] = h_last @ fc_w^T + fc_b.   64 WMMA tiles, 8 blocks.
// ---------------------------------------------------------------------------
__global__ __launch_bounds__(256)
void fc_kernel(const bf16_t* __restrict__ hlast,
               const bf16_t* __restrict__ Wfc,
               const float*  __restrict__ bias,
               float*        __restrict__ out)
{
    const int lane = threadIdx.x & 31;
    const int wid  = blockIdx.x * 8 + (threadIdx.x >> 5); // 0..63
    const int mt   = wid >> 4;  // 0..3
    const int nt   = wid & 15;  // 0..15

    const int m   = lane & 15;
    const int kA0 = (lane < 16) ? 0 : 16;
    const int nB  = lane & 15;
    const int kB0 = (lane < 16) ? 0 : 16;

    v8f acc = {};
    const bf16_t* arow = hlast + (size_t)(mt * 16 + m) * H;
    const bf16_t* brow = Wfc + (size_t)(nt * 16 + nB) * H;
    for (int kk = 0; kk < H; kk += 32) {
        v16bf a = ld_frag(arow + kk + kA0, arow + kk + kA0 + 8);
        v16bf b = ld_frag(brow + kk + kB0, brow + kk + kB0 + 8);
        acc = wmma_bf16(a, b, acc);
    }
    const int mo = (lane < 16) ? 0 : 8;
    const float bv = bias[nt * 16 + (lane & 15)];
#pragma unroll
    for (int r = 0; r < 8; ++r) {
        out[(size_t)(mt * 16 + mo + r) * O + nt * 16 + (lane & 15)] = acc[r] + bv;
    }
}

// ---------------------------------------------------------------------------
// Utility kernels
// ---------------------------------------------------------------------------
__global__ void f32_to_bf16_kernel(const float* __restrict__ src,
                                   bf16_t* __restrict__ dst, size_t n) {
    size_t i      = (size_t)blockIdx.x * blockDim.x + threadIdx.x;
    size_t stride = (size_t)gridDim.x * blockDim.x;
    for (; i < n; i += stride) dst[i] = (bf16_t)src[i];
}

__global__ void bias_sum_kernel(const float* __restrict__ a,
                                const float* __restrict__ b,
                                float* __restrict__ o, size_t n) {
    size_t i      = (size_t)blockIdx.x * blockDim.x + threadIdx.x;
    size_t stride = (size_t)gridDim.x * blockDim.x;
    for (; i < n; i += stride) o[i] = a[i] + b[i];
}

__global__ void zero_u32_kernel(unsigned int* __restrict__ p, size_t n) {
    size_t i      = (size_t)blockIdx.x * blockDim.x + threadIdx.x;
    size_t stride = (size_t)gridDim.x * blockDim.x;
    for (; i < n; i += stride) p[i] = 0u;
}

// ---------------------------------------------------------------------------
extern "C" void kernel_launch(void* const* d_in, const int* in_sizes, int n_in,
                              void* d_out, int out_size, void* d_ws, size_t ws_size,
                              hipStream_t stream) {
    (void)in_sizes; (void)n_in; (void)out_size;

    const float* x    = (const float*)d_in[0]; // [B,T,D]
    const float* Wih  = (const float*)d_in[1]; // [L,4H,D]
    const float* Whh  = (const float*)d_in[2]; // [L,4H,H]
    const float* bih  = (const float*)d_in[3]; // [L,4H]
    const float* bhh  = (const float*)d_in[4]; // [L,4H]
    const float* fcw  = (const float*)d_in[5]; // [O,H]
    const float* fcb  = (const float*)d_in[6]; // [O]
    float* out        = (float*)d_out;         // [B,O]

    // workspace carve-up (256B aligned)
    char* base = (char*)d_ws;
    size_t off = 0;
    auto carve = [&](size_t bytes) -> char* {
        char* p = base + off;
        off = (off + bytes + 255) & ~(size_t)255;
        return p;
    };
    const size_t nWih  = (size_t)L * FOURH * D;   // 16M
    const size_t nWhh  = (size_t)L * FOURH * H;   // 16M
    const size_t nX    = (size_t)B * T * D;       // 33.5M
    const size_t nFcw  = (size_t)O * H;
    const size_t nBias = (size_t)L * FOURH;

    bf16_t* wih_bf = (bf16_t*)carve(nWih * sizeof(bf16_t));
    bf16_t* whh_bf = (bf16_t*)carve(nWhh * sizeof(bf16_t));
    bf16_t* x_bf   = (bf16_t*)carve(nX * sizeof(bf16_t));
    bf16_t* fcw_bf = (bf16_t*)carve(nFcw * sizeof(bf16_t));
    float*  bsum   = (float*)carve(nBias * sizeof(float));
    // h double buffer [2][L][B][H] bf16 followed immediately by c [L][B][H] f32
    bf16_t* hbuf   = (bf16_t*)carve(2ull * L * B * H * sizeof(bf16_t));
    float*  cbuf   = (float*)carve((size_t)L * B * H * sizeof(float));
    // optional hoisted layer-0 input projection: [T,B,4H] f32 (512 MB)
    float*  xg     = (float*)carve((size_t)T * B * FOURH * sizeof(float));
    const bool use_xg = (off <= ws_size);   // ws_size is fixed -> deterministic

    // one-time conversions / init
    f32_to_bf16_kernel<<<4096, 256, 0, stream>>>(Wih, wih_bf, nWih);
    f32_to_bf16_kernel<<<4096, 256, 0, stream>>>(Whh, whh_bf, nWhh);
    f32_to_bf16_kernel<<<4096, 256, 0, stream>>>(x, x_bf, nX);
    f32_to_bf16_kernel<<<512, 256, 0, stream>>>(fcw, fcw_bf, nFcw);
    bias_sum_kernel<<<64, 256, 0, stream>>>(bih, bhh, bsum, nBias);
    {
        // zero h (both parities, bf16) and c (f32) — contiguous region
        size_t bytes = 2ull * L * B * H * sizeof(bf16_t) + (size_t)L * B * H * sizeof(float);
        zero_u32_kernel<<<2048, 256, 0, stream>>>((unsigned int*)hbuf, bytes / 4);
    }

    // hoisted, fully parallel layer-0 input projection over all timesteps
    if (use_xg) {
        xproj_kernel<<<dim3(32, T), 512, 0, stream>>>(x_bf, wih_bf, xg);
    }

    // recurrence: one fused WMMA+cell kernel per (t, layer)
    for (int t = 0; t < T; ++t) {
        const int p = t & 1;
        for (int l = 0; l < L; ++l) {
            const bf16_t* inA;
            unsigned long long ldA;
            const float* xgate = nullptr;
            if (l == 0) {
                if (use_xg) {
                    inA = nullptr;
                    ldA = 0;
                    xgate = xg + (size_t)t * B * FOURH;
                } else {
                    inA = x_bf + (size_t)t * D;        // row b at b*T*D
                    ldA = (unsigned long long)T * D;
                }
            } else {
                inA = hbuf + ((size_t)p * L + (l - 1)) * B * H; // layer below, this step
                ldA = H;
            }
            const bf16_t* hprev = hbuf + ((size_t)(1 - p) * L + l) * B * H;
            bf16_t* hout        = hbuf + ((size_t)p * L + l) * B * H;
            lstm_step_kernel<<<32, 512, 0, stream>>>(
                inA, ldA, xgate, hprev,
                wih_bf + (size_t)l * FOURH * D,
                whh_bf + (size_t)l * FOURH * H,
                bsum + (size_t)l * FOURH,
                cbuf + (size_t)l * B * H,
                hout);
        }
    }

    // final projection from last layer's h at t = T-1
    const int plast = (T - 1) & 1;
    const bf16_t* hlast = hbuf + ((size_t)plast * L + (L - 1)) * B * H;
    fc_kernel<<<8, 256, 0, stream>>>(hlast, fcw_bf, fcb, out);
}
